// VectorQuantizerEMA_6597069767077
// MI455X (gfx1250) — matrix operand order repaired and checked by reading, not compiled, not run on hardware
//
#include <hip/hip_runtime.h>
#include <math.h>

typedef __attribute__((ext_vector_type(16))) __bf16 v16bf;
typedef __attribute__((ext_vector_type(8)))  float  v8f;

#define NUM_EMBED 1024
#define EMBED_DIM 64
#define NROWS     65536      // 16*64*64 vectors
#define ZQ_ELEMS  4194304    // 16*64*64*64
#define HWSZ      4096       // 64*64

// workspace layout (bytes)
#define WS_B_USHORTS (2*64*2*32*16)           // part x ct x kt x lane x 16 bf16
#define WS_HIST_OFF  (WS_B_USHORTS*2)         // 262144
#define WS_LOSS_OFF  (WS_HIST_OFF + 1024*4)

__device__ __forceinline__ unsigned short f2bf(float f){
  unsigned u = __builtin_bit_cast(unsigned, f);
  unsigned r = u + 0x7FFFu + ((u >> 16) & 1u);   // round to nearest even
  return (unsigned short)(r >> 16);
}
__device__ __forceinline__ float bf2f(unsigned short h){
  unsigned u = ((unsigned)h) << 16;
  return __builtin_bit_cast(float, u);
}
__device__ __forceinline__ __bf16 bfb(unsigned short h){
  return __builtin_bit_cast(__bf16, h);
}
__device__ __forceinline__ v8f wmma_bf16(v16bf a, v16bf b, v8f c){
  // D = A(16x32 bf16) * B(32x16 bf16) + C(16x16 f32)
  return __builtin_amdgcn_wmma_f32_16x16x32_bf16(false, a, false, b, (short)0, c, false, false);
}

__global__ void vq_zero(unsigned* __restrict__ hist, float* __restrict__ lossAcc){
  hist[threadIdx.x] = 0u;
  if (threadIdx.x == 0) *lossAcc = 0.f;
}

// Normalize codebook, split to bf16 hi/lo, pack into WMMA B layout:
// B[k][n]: lane = n + 16*h, element i -> K = i + 8*h + 8*(i>=8) within 32-chunk.
__global__ void vq_prep(const float* __restrict__ embed, unsigned short* __restrict__ wsB){
  int k = blockIdx.x * blockDim.x + threadIdx.x;
  if (k >= NUM_EMBED) return;
  const float* row = embed + k * EMBED_DIM;
  float v[EMBED_DIM];
  float ss = 0.f;
#pragma unroll
  for (int d = 0; d < EMBED_DIM; ++d){ v[d] = row[d]; ss += v[d]*v[d]; }
  float inv = 1.0f / fmaxf(sqrtf(ss), 1e-12f);
  int ct = k >> 4, nc = k & 15;
#pragma unroll
  for (int d = 0; d < EMBED_DIM; ++d){
    float zn = v[d] * inv;
    unsigned short hi = f2bf(zn);
    unsigned short lo = f2bf(zn - bf2f(hi));
    int kt = d >> 5, kl = d & 31;
    int lane = nc + (((kl >> 3) & 1) << 4);
    int i    = (kl & 7) + ((kl >> 4) << 3);
    wsB[(((( (0*64) + ct)*2 + kt)*32 + lane)*16 + i)] = hi;
    wsB[(((( (1*64) + ct)*2 + kt)*32 + lane)*16 + i)] = lo;
  }
}

// One wave32 per 32 rows. Phase 1: pack A (hi/lo). Phase 2: 64 col tiles of WMMA
// + running argmax. Phase 3: gather z_q, write out, loss + histogram.
__global__ __launch_bounds__(32) void vq_main(const float* __restrict__ z,
                                              const float* __restrict__ embed,
                                              const unsigned short* __restrict__ wsB,
                                              float* __restrict__ zq_out,
                                              float* __restrict__ idx_out,
                                              unsigned* __restrict__ hist,
                                              float* __restrict__ lossAcc)
{
  __shared__ int sIdx[32];
  const int lane = threadIdx.x;
  const int h    = lane >> 4;     // which 16-lane half
  const int r    = lane & 15;
  const int n0   = blockIdx.x * 32;

  // A operands: [tile][kt], 16-bit A layout: lane (h,r) row m=r, element i -> k = i + 8h + 8*(i>=8)
  v16bf ah[2][2], al[2][2];

#pragma unroll
  for (int t = 0; t < 2; ++t){
    int n  = n0 + t*16 + r;
    int b  = n >> 12;
    int hw = n & 4095;
    long base = (long)b * 262144 + hw;        // z[b][d][hw] = base + d*4096
    float v[4][8];
    float ss = 0.f;
#pragma unroll
    for (int g = 0; g < 4; ++g)
#pragma unroll
      for (int c8 = 0; c8 < 8; ++c8){
        int d = 8*h + c8 + 16*g;              // this lane's half of the 64 dims
        float x = z[base + (long)d * 4096];
        v[g][c8] = x;
        ss += x * x;
      }
    ss += __shfl_xor(ss, 16, 32);             // combine halves of the row
    float inv = 1.0f / fmaxf(sqrtf(ss), 1e-12f);
#pragma unroll
    for (int g = 0; g < 4; ++g)
#pragma unroll
      for (int c8 = 0; c8 < 8; ++c8){
        float zn = v[g][c8] * inv;
        unsigned short hi = f2bf(zn);
        unsigned short lo = f2bf(zn - bf2f(hi));
        int kt = g >> 1;
        int i  = c8 + 8*(g & 1);              // d = 8h + c8 + 16g  ->  k = i + 8h + 8*(i>=8)
        ah[t][kt][i] = bfb(hi);
        al[t][kt][i] = bfb(lo);
      }
  }

  float best0[8], best1[8]; int bi0[8], bi1[8];
#pragma unroll
  for (int vv = 0; vv < 8; ++vv){ best0[vv] = -3.4e38f; best1[vv] = -3.4e38f; bi0[vv] = 0; bi1[vv] = 0; }

  const v16bf* bp = reinterpret_cast<const v16bf*>(wsB);
  for (int ct = 0; ct < 64; ++ct){
    v16bf bh0 = bp[(((0*64 + ct)*2 + 0)*32 + lane)];
    v16bf bh1 = bp[(((0*64 + ct)*2 + 1)*32 + lane)];
    v16bf bl0 = bp[(((1*64 + ct)*2 + 0)*32 + lane)];
    v16bf bl1 = bp[(((1*64 + ct)*2 + 1)*32 + lane)];
    v8f c0 = {}; v8f c1 = {};
    // hi*hi + hi*lo + lo*hi  (fp32 accumulate) ~ fp32-accurate similarity
    c0 = wmma_bf16(ah[0][0], bh0, c0);  c1 = wmma_bf16(ah[1][0], bh0, c1);
    c0 = wmma_bf16(ah[0][1], bh1, c0);  c1 = wmma_bf16(ah[1][1], bh1, c1);
    c0 = wmma_bf16(ah[0][0], bl0, c0);  c1 = wmma_bf16(ah[1][0], bl0, c1);
    c0 = wmma_bf16(ah[0][1], bl1, c0);  c1 = wmma_bf16(ah[1][1], bl1, c1);
    c0 = wmma_bf16(al[0][0], bh0, c0);  c1 = wmma_bf16(al[1][0], bh0, c1);
    c0 = wmma_bf16(al[0][1], bh1, c0);  c1 = wmma_bf16(al[1][1], bh1, c1);
    int cc = ct*16 + r;                       // this lane's codebook column
#pragma unroll
    for (int vv = 0; vv < 8; ++vv){           // D layout: lane=col, VGPR vv -> row vv+8h
      if (c0[vv] > best0[vv]){ best0[vv] = c0[vv]; bi0[vv] = cc; }
      if (c1[vv] > best1[vv]){ best1[vv] = c1[vv]; bi1[vv] = cc; }
    }
  }

  // argmax reduce across the 16 lanes of each half; tie -> smaller index (jnp.argmax)
#pragma unroll
  for (int vv = 0; vv < 8; ++vv){
    for (int m = 1; m < 16; m <<= 1){
      float ov = __shfl_xor(best0[vv], m, 32); int oi = __shfl_xor(bi0[vv], m, 32);
      if (ov > best0[vv] || (ov == best0[vv] && oi < bi0[vv])){ best0[vv] = ov; bi0[vv] = oi; }
      float ow = __shfl_xor(best1[vv], m, 32); int oj = __shfl_xor(bi1[vv], m, 32);
      if (ow > best1[vv] || (ow == best1[vv] && oj < bi1[vv])){ best1[vv] = ow; bi1[vv] = oj; }
    }
  }
  if (r == 0){                                // lanes 0 and 16: rows 8h..8h+7 of each tile
#pragma unroll
    for (int vv = 0; vv < 8; ++vv){
      int m0 = 8*h + vv;
      sIdx[m0]      = bi0[vv];
      sIdx[16 + m0] = bi1[vv];
      idx_out[n0 + m0]      = (float)bi0[vv];
      idx_out[n0 + 16 + m0] = (float)bi1[vv];
      atomicAdd(&hist[bi0[vv]], 1u);
      atomicAdd(&hist[bi1[vv]], 1u);
    }
  }
  __syncthreads();

  // z_q gather/scatter + loss (uses ORIGINAL codebook and ORIGINAL z)
  float lsum = 0.f;
#pragma unroll
  for (int t = 0; t < 2; ++t){
    int n   = n0 + t*16 + r;
    int idx = sIdx[t*16 + r];
    int b   = n >> 12;
    int hw  = n & 4095;
    long base = (long)b * 262144 + hw;
    const float* erow = embed + idx * EMBED_DIM;
#pragma unroll
    for (int g = 0; g < 4; ++g)
#pragma unroll
      for (int c8 = 0; c8 < 8; ++c8){
        int d = 8*h + c8 + 16*g;
        float e = erow[d];
        float x = z[base + (long)d * 4096];
        zq_out[base + (long)d * 4096] = e;    // straight-through fwd == z_q
        float df = e - x;
        lsum += df * df;
      }
  }
  for (int m = 1; m < 32; m <<= 1) lsum += __shfl_xor(lsum, m, 32);
  if (lane == 0) atomicAdd(lossAcc, lsum);
}

__global__ __launch_bounds__(32) void vq_final(const unsigned* __restrict__ hist,
                                               const float* __restrict__ lossAcc,
                                               float* __restrict__ tail)
{
  int lane = threadIdx.x;
  float s = 0.f;
  for (int b = lane; b < NUM_EMBED; b += 32){
    float p = (float)hist[b] * (1.0f / (float)NROWS);
    s += p * logf(p + 1e-10f);
  }
  for (int m = 1; m < 32; m <<= 1) s += __shfl_xor(s, m, 32);
  if (lane == 0){
    tail[0] = 1.25f * (*lossAcc) * (1.0f / (float)ZQ_ELEMS);  // (BETA+1)*MSE
    tail[1] = expf(-s);
  }
}

extern "C" void kernel_launch(void* const* d_in, const int* in_sizes, int n_in,
                              void* d_out, int out_size, void* d_ws, size_t ws_size,
                              hipStream_t stream)
{
  const float* z     = (const float*)d_in[0];   // [16,64,64,64]
  const float* embed = (const float*)d_in[1];   // [1024,64]
  float* out     = (float*)d_out;
  float* zq_out  = out;                         // [16,64,64,64]
  float* idx_out = out + ZQ_ELEMS;              // [65536] (indices as float)
  float* tail    = out + ZQ_ELEMS + NROWS;      // loss, perplexity

  unsigned short* wsB = (unsigned short*)d_ws;
  unsigned* hist  = (unsigned*)((char*)d_ws + WS_HIST_OFF);
  float* lossAcc  = (float*)((char*)d_ws + WS_LOSS_OFF);

  vq_zero <<<1, 1024, 0, stream>>>(hist, lossAcc);
  vq_prep <<<4, 256,  0, stream>>>(embed, wsB);
  vq_main <<<NROWS/32, 32, 0, stream>>>(z, embed, wsB, zq_out, idx_out, hist, lossAcc);
  vq_final<<<1, 32,   0, stream>>>(hist, lossAcc, tail);
}
